// MyAttention_69784628625564
// MI455X (gfx1250) — compile-verified, matrix operand order checked by reading, not run on hardware
//
#include <hip/hip_runtime.h>
#include <hip/hip_bf16.h>
#include <math.h>

typedef __attribute__((ext_vector_type(16))) _Float16 v16h;
typedef __attribute__((ext_vector_type(8)))  _Float16 v8h;
typedef __attribute__((ext_vector_type(8)))  float    v8f;

union V16H { v16h v; v8h h[2]; _Float16 e[16]; };

#define BATCH 2
#define NTOK  8192
#define CDIM  256
#define NHEAD 8
#define DH    32
#define HMAP  128
#define WMAP  128
#define HWMAP (HMAP*WMAP)
#define NS    1024          // (128/4)^2
#define KCONV 4096          // C*4*4
#define GEMM_NB 4           // N-tiles per wave (A-operand reuse)

// ---- helpers -------------------------------------------------------------

// Load 32 consecutive K-elements of one row into the CDNA5 16-bit A/B operand
// layout: lane (row, half) holds K = [half*8, half*8+8) ++ [16+half*8, 16+half*8+8)
__device__ __forceinline__ v16h load_k32(const _Float16* p, int half) {
  V16H r;
  r.h[0] = *(const v8h*)(p + half * 8);
  r.h[1] = *(const v8h*)(p + 16 + half * 8);
  return r.v;
}

// ---- elementwise / preprocessing kernels ---------------------------------

__global__ void cvt_f32_f16_kernel(const float* __restrict__ in,
                                   _Float16* __restrict__ out, size_t n) {
  size_t i = blockIdx.x * (size_t)blockDim.x + threadIdx.x;
  size_t stride = (size_t)gridDim.x * blockDim.x;
  for (; i < n; i += stride) out[i] = (_Float16)in[i];
}

// token -> pixel index (round-half-even like jnp.round)
__global__ void pix_kernel(const float* __restrict__ loc, int* __restrict__ pix) {
  int t = blockIdx.x * blockDim.x + threadIdx.x;
  if (t >= BATCH * NTOK) return;
  float lx = loc[(size_t)t * 2 + 0], ly = loc[(size_t)t * 2 + 1];
  lx = fminf(fmaxf(lx, 0.f), 1.f) * (float)(WMAP - 1);
  ly = fminf(fmaxf(ly, 0.f), 1.f) * (float)(HMAP - 1);
  pix[t] = (int)rintf(ly) * WMAP + (int)rintf(lx);
}

// one thread per (token, channel incl. count channel): atomic scatter
__global__ void scatter_kernel(const float* __restrict__ xsrc,
                               const int* __restrict__ pix,
                               float* __restrict__ map) {
  size_t tid = blockIdx.x * (size_t)blockDim.x + threadIdx.x;
  size_t total = (size_t)BATCH * NTOK * (CDIM + 1);
  if (tid >= total) return;
  int c = (int)(tid % (CDIM + 1));
  size_t tok = tid / (CDIM + 1);
  int b = (int)(tok / NTOK);
  int p = pix[tok];
  float val = (c < CDIM) ? xsrc[tok * CDIM + c] : 1.0f;
  atomicAdd(&map[((size_t)b * (CDIM + 1) + c) * HWMAP + p], val);
}

// feature /= (count + 1e-6); zero where count==0 (count channel left raw)
__global__ void norm_map_kernel(float* __restrict__ map) {
  size_t tid = blockIdx.x * (size_t)blockDim.x + threadIdx.x;
  size_t total = (size_t)BATCH * CDIM * HWMAP;
  if (tid >= total) return;
  int pix = (int)(tid % HWMAP);
  size_t bc = tid / HWMAP;
  int c = (int)(bc % CDIM);
  int b = (int)(bc / CDIM);
  float cnt = map[((size_t)b * (CDIM + 1) + CDIM) * HWMAP + pix];
  size_t off = ((size_t)b * (CDIM + 1) + c) * HWMAP + pix;
  float f = map[off] / (cnt + 1e-6f);
  map[off] = (cnt > 0.f) ? f : 0.f;
}

// 3x3 gaussian (sigma=2, kernel-normalized, zero-pad) + masked reconstruct
__global__ void reconstruct_kernel(const float* __restrict__ map,
                                   float* __restrict__ out) {
  size_t tid = blockIdx.x * (size_t)blockDim.x + threadIdx.x;
  size_t total = (size_t)BATCH * CDIM * HWMAP;
  if (tid >= total) return;
  int pix = (int)(tid % HWMAP);
  size_t bc = tid / HWMAP;
  int c = (int)(bc % CDIM);
  int b = (int)(bc / CDIM);
  int y = pix / WMAP, x = pix % WMAP;
  const float* featc = map + ((size_t)b * (CDIM + 1) + c) * HWMAP;
  const float* cnt   = map + ((size_t)b * (CDIM + 1) + CDIM) * HWMAP;
  float gf = 0.f, gm = 0.f, gsum = 0.f;
#pragma unroll
  for (int dy = -1; dy <= 1; ++dy) {
#pragma unroll
    for (int dx = -1; dx <= 1; ++dx) {
      float g = __expf(-(float)(dx * dx + dy * dy) * 0.125f); // /(2*sigma^2)=1/8
      gsum += g;
      int yy = y + dy, xx = x + dx;
      if (yy < 0 || yy >= HMAP || xx < 0 || xx >= WMAP) continue;
      int q = yy * WMAP + xx;
      float mb = (cnt[q] > 0.f) ? 1.f : 0.f;
      gf += g * featc[q] * mb;
      gm += g * mb;
    }
  }
  gf /= gsum; gm /= gsum;
  float f_int = gf / (gm + 1e-6f);
  f_int = (gm > 0.f) ? f_int : 0.f;
  float mbc = (cnt[pix] > 0.f) ? 1.f : 0.f;
  out[((size_t)b * CDIM + c) * HWMAP + pix] = featc[pix] + (1.f - mbc) * f_int;
}

// stride-4 4x4 conv patches -> rows of [B*NS, 4096] f16 (pure gather)
__global__ void im2col_kernel(const float* __restrict__ map2,
                              _Float16* __restrict__ out) {
  size_t tid = blockIdx.x * (size_t)blockDim.x + threadIdx.x;
  size_t total = (size_t)BATCH * NS * KCONV;
  if (tid >= total) return;
  int col = (int)(tid & (KCONV - 1));
  size_t rowi = tid >> 12;
  int b = (int)(rowi / NS);
  int p = (int)(rowi % NS);
  int oy = p >> 5, ox = p & 31;
  int i = col >> 4;
  int kh = (col >> 2) & 3;
  int kw = col & 3;
  int yy = oy * 4 + kh, xx = ox * 4 + kw;
  out[tid] = (_Float16)map2[(((size_t)b * CDIM + i) * HMAP + yy) * WMAP + xx];
}

// pre-swizzle V into WMMA B-operand layout:
// Vsw[b][h][kchunk][g][lane][ee], value = V[key(lane,ee)][h*32 + g*16 + (lane&15)]
__global__ void vswz_kernel(const _Float16* __restrict__ Vh,
                            _Float16* __restrict__ out) {
  size_t tid = blockIdx.x * (size_t)blockDim.x + threadIdx.x;
  size_t total = (size_t)BATCH * NHEAD * (NS / 32) * 2 * 32 * 16;
  if (tid >= total) return;
  int ee  = (int)(tid & 15);
  int L   = (int)(tid >> 4) & 31;
  int g   = (int)(tid >> 9) & 1;
  int kcb = (int)(tid >> 10) & 31;
  int h   = (int)(tid >> 15) & 7;
  int b   = (int)(tid >> 18);
  int half = L >> 4, r = L & 15;
  int jj = ee >> 1, rr = ee & 1;
  int key = kcb * 32 + ((jj >= 4) ? 16 : 0) + half * 8 + ((jj & 3) << 1) + rr;
  out[tid] = Vh[((size_t)(b * NS + key)) * CDIM + h * DH + g * 16 + r];
}

// row-wise LayerNorm over C=256, emits f16
__global__ __launch_bounds__(CDIM) void ln_kernel(const float* __restrict__ xin,
                                                  const float* __restrict__ w,
                                                  const float* __restrict__ bb,
                                                  _Float16* __restrict__ out) {
  __shared__ float red[CDIM];
  int rowi = blockIdx.x, t = threadIdx.x;
  float v = xin[(size_t)rowi * CDIM + t];
  red[t] = v; __syncthreads();
  for (int s = CDIM / 2; s > 0; s >>= 1) { if (t < s) red[t] += red[t + s]; __syncthreads(); }
  float mu = red[0] * (1.0f / CDIM);
  __syncthreads();
  float d = v - mu;
  red[t] = d * d; __syncthreads();
  for (int s = CDIM / 2; s > 0; s >>= 1) { if (t < s) red[t] += red[t + s]; __syncthreads(); }
  float var = red[0] * (1.0f / CDIM);
  float y = d * rsqrtf(var + 1e-5f) * w[t] + bb[t];
  out[(size_t)rowi * CDIM + t] = (_Float16)y;
}

// ---- WMMA GEMM: C[M,N] = A[M,K] @ Bm[N,K]^T (+bias), f16 in ---------------
// one wave computes 16 x (16*GEMM_NB): A operand reused across 4 WMMAs/k-step
__global__ __launch_bounds__(256) void gemm_f16_wmma(
    const _Float16* __restrict__ A, const _Float16* __restrict__ Bm,
    const float* __restrict__ bias, float* __restrict__ Cf,
    _Float16* __restrict__ Ch, int M, int N, int K) {
  int tileM = blockIdx.x * blockDim.y + threadIdx.y;   // wave-uniform
  if (tileM * 16 >= M) return;
  int ng = blockIdx.y;                                 // group of GEMM_NB N-tiles
  int lane = threadIdx.x;
  int r = lane & 15, half = lane >> 4;
  const _Float16* arow = A + (size_t)(tileM * 16 + r) * K;
  const _Float16* brow = Bm + (size_t)(ng * (16 * GEMM_NB) + r) * K;
  v8f acc[GEMM_NB] = {};
  for (int k0 = 0; k0 < K; k0 += 32) {
    v16h a = load_k32(arow + k0, half);
#pragma unroll
    for (int nt = 0; nt < GEMM_NB; ++nt) {
      v16h b = load_k32(brow + (size_t)nt * 16 * K + k0, half);
      acc[nt] = __builtin_amdgcn_wmma_f32_16x16x32_f16(false, a, false, b,
                                                       (short)0, acc[nt], false, false);
    }
  }
  int rbase = tileM * 16 + half * 8;
#pragma unroll
  for (int nt = 0; nt < GEMM_NB; ++nt) {
    int col = ng * (16 * GEMM_NB) + nt * 16 + r;
    float bv = bias ? bias[col] : 0.f;
#pragma unroll
    for (int i = 0; i < 8; ++i) {
      float v = acc[nt][i] + bv;
      size_t off = (size_t)(rbase + i) * N + col;
      if (Cf) Cf[off] = v;
      if (Ch) Ch[off] = (_Float16)v;
    }
  }
}

// ---- attention: one wave per (b, h, 16-query tile), Ns=1024 keys ----------
__global__ __launch_bounds__(32) void attn_kernel(
    const _Float16* __restrict__ Q, const _Float16* __restrict__ Kt,
    const _Float16* __restrict__ Vsw, const float* __restrict__ conf,
    _Float16* __restrict__ ctx) {
  __shared__ __align__(16) _Float16 P[16 * NS];   // 32 KB
  int bh = blockIdx.y;
  int b = bh >> 3, h = bh & 7;
  int qtile = blockIdx.x;
  int lane = threadIdx.x;
  int r = lane & 15, half = lane >> 4;
  const float scale = 0.17677669529663687f;       // 1/sqrt(32)

  const _Float16* qrow = Q + ((size_t)(b * NTOK + qtile * 16 + r)) * CDIM + h * DH;
  v16h aq = load_k32(qrow, half);                 // K = dh = 32: one WMMA per S tile

  // Phase 1: S = scale * Q K^T + conf  -> LDS (f16, raw scores)
  for (int j = 0; j < NS / 16; ++j) {
    const _Float16* krow = Kt + ((size_t)(b * NS + j * 16 + r)) * CDIM + h * DH;
    v16h bk = load_k32(krow, half);
    v8f s = {};
    s = __builtin_amdgcn_wmma_f32_16x16x32_f16(false, aq, false, bk,
                                               (short)0, s, false, false);
    float cb = conf[(size_t)b * NS + j * 16 + r]; // this lane's key column
#pragma unroll
    for (int i = 0; i < 8; ++i)
      P[(half * 8 + i) * NS + j * 16 + r] = (_Float16)(s[i] * scale + cb);
  }
  __syncthreads();

  // Phase 2: softmax; 2 lanes per row (half splits the 1024 columns)
  {
    float mx = -1e30f;
    int c0 = half * (NS / 2);
    for (int c = c0; c < c0 + NS / 2; ++c)
      mx = fmaxf(mx, (float)P[r * NS + c]);
    mx = fmaxf(mx, __shfl_xor(mx, 16, 32));
    float sum = 0.f;
    for (int c = c0; c < c0 + NS / 2; ++c)
      sum += __expf((float)P[r * NS + c] - mx);
    sum += __shfl_xor(sum, 16, 32);
    float inv = 1.0f / sum;
    for (int c = c0; c < c0 + NS / 2; ++c)
      P[r * NS + c] = (_Float16)(__expf((float)P[r * NS + c] - mx) * inv);
  }
  __syncthreads();

  // Phase 3: O = P @ V; V pre-swizzled -> two 16B vector loads per operand
  v8f acc0 = {}, acc1 = {};
  for (int kcb = 0; kcb < NS / 32; ++kcb) {
    const _Float16* prow = &P[r * NS + kcb * 32];
    V16H pa;
    pa.h[0] = *(const v8h*)(prow + half * 8);
    pa.h[1] = *(const v8h*)(prow + 16 + half * 8);
    const _Float16* vbase =
        Vsw + (((size_t)(b * NHEAD + h) * (NS / 32) + kcb) * 2 * 32) * 16;
    V16H vb0, vb1;
    vb0.h[0] = *(const v8h*)(vbase + lane * 16);
    vb0.h[1] = *(const v8h*)(vbase + lane * 16 + 8);
    vb1.h[0] = *(const v8h*)(vbase + 512 + lane * 16);
    vb1.h[1] = *(const v8h*)(vbase + 512 + lane * 16 + 8);
    acc0 = __builtin_amdgcn_wmma_f32_16x16x32_f16(false, pa.v, false, vb0.v,
                                                  (short)0, acc0, false, false);
    acc1 = __builtin_amdgcn_wmma_f32_16x16x32_f16(false, pa.v, false, vb1.v,
                                                  (short)0, acc1, false, false);
  }
#pragma unroll
  for (int i = 0; i < 8; ++i) {
    int m = qtile * 16 + half * 8 + i;
    _Float16* op = ctx + ((size_t)(b * NTOK + m)) * CDIM + h * DH;
    op[r] = (_Float16)acc0[i];
    op[r + 16] = (_Float16)acc1[i];
  }
}

// ---- host side ------------------------------------------------------------

extern "C" void kernel_launch(void* const* d_in, const int* in_sizes, int n_in,
                              void* d_out, int out_size, void* d_ws, size_t ws_size,
                              hipStream_t stream) {
  const float* x      = (const float*)d_in[0];
  const float* xsrc   = (const float*)d_in[1];
  const float* loc    = (const float*)d_in[2];
  const float* conf   = (const float*)d_in[3];
  const float* Wq     = (const float*)d_in[4];
  const float* Wk     = (const float*)d_in[5];
  const float* Wv     = (const float*)d_in[6];
  const float* sr_w   = (const float*)d_in[7];
  const float* sr_b   = (const float*)d_in[8];
  const float* ln_w   = (const float*)d_in[9];
  const float* ln_b   = (const float*)d_in[10];
  const float* Wp     = (const float*)d_in[11];
  const float* bp     = (const float*)d_in[12];

  char* w = (char*)d_ws;
  size_t o = 0;
  auto take = [&](size_t bytes) { char* p = w + o; o += (bytes + 255) & ~(size_t)255; return p; };
  float*    map   = (float*)take((size_t)BATCH * (CDIM + 1) * HWMAP * 4);
  float*    map2  = (float*)take((size_t)BATCH * CDIM * HWMAP * 4);
  _Float16* xh    = (_Float16*)take((size_t)BATCH * NTOK * CDIM * 2);
  _Float16* wqh   = (_Float16*)take((size_t)CDIM * CDIM * 2);
  _Float16* wkh   = (_Float16*)take((size_t)CDIM * CDIM * 2);
  _Float16* wvh   = (_Float16*)take((size_t)CDIM * CDIM * 2);
  _Float16* wph   = (_Float16*)take((size_t)CDIM * CDIM * 2);
  _Float16* srwh  = (_Float16*)take((size_t)CDIM * KCONV * 2);
  _Float16* im2c  = (_Float16*)take((size_t)BATCH * NS * KCONV * 2);
  float*    xspre = (float*)take((size_t)BATCH * NS * CDIM * 4);
  _Float16* xsh   = (_Float16*)take((size_t)BATCH * NS * CDIM * 2);
  _Float16* Qh    = (_Float16*)take((size_t)BATCH * NTOK * CDIM * 2);
  _Float16* Kh    = (_Float16*)take((size_t)BATCH * NS * CDIM * 2);
  _Float16* Vh    = (_Float16*)take((size_t)BATCH * NS * CDIM * 2);
  _Float16* Vsw   = (_Float16*)take((size_t)BATCH * NS * CDIM * 2);
  _Float16* ctxh  = (_Float16*)take((size_t)BATCH * NTOK * CDIM * 2);
  int*      pixb  = (int*)take((size_t)BATCH * NTOK * 4);

  auto nblk = [](size_t n, int t) { return (unsigned)((n + (size_t)t - 1) / t); };

  // f16 conversions
  cvt_f32_f16_kernel<<<4096, 256, 0, stream>>>(x,    xh,   (size_t)BATCH * NTOK * CDIM);
  cvt_f32_f16_kernel<<<256,  256, 0, stream>>>(Wq,   wqh,  (size_t)CDIM * CDIM);
  cvt_f32_f16_kernel<<<256,  256, 0, stream>>>(Wk,   wkh,  (size_t)CDIM * CDIM);
  cvt_f32_f16_kernel<<<256,  256, 0, stream>>>(Wv,   wvh,  (size_t)CDIM * CDIM);
  cvt_f32_f16_kernel<<<256,  256, 0, stream>>>(Wp,   wph,  (size_t)CDIM * CDIM);
  cvt_f32_f16_kernel<<<2048, 256, 0, stream>>>(sr_w, srwh, (size_t)CDIM * KCONV);

  // token2map
  (void)hipMemsetAsync(map, 0, (size_t)BATCH * (CDIM + 1) * HWMAP * 4, stream);
  pix_kernel<<<nblk((size_t)BATCH * NTOK, 256), 256, 0, stream>>>(loc, pixb);
  scatter_kernel<<<nblk((size_t)BATCH * NTOK * (CDIM + 1), 256), 256, 0, stream>>>(xsrc, pixb, map);
  norm_map_kernel<<<nblk((size_t)BATCH * CDIM * HWMAP, 256), 256, 0, stream>>>(map);
  reconstruct_kernel<<<nblk((size_t)BATCH * CDIM * HWMAP, 256), 256, 0, stream>>>(map, map2);

  // conv (stride4 4x4) as GEMM
  im2col_kernel<<<nblk((size_t)BATCH * NS * KCONV, 256), 256, 0, stream>>>(map2, im2c);
  {
    dim3 blk(32, 8), grd((BATCH * NS / 16) / 8, CDIM / (16 * GEMM_NB));
    gemm_f16_wmma<<<grd, blk, 0, stream>>>(im2c, srwh, sr_b, xspre, nullptr,
                                           BATCH * NS, CDIM, KCONV);
  }
  ln_kernel<<<BATCH * NS, CDIM, 0, stream>>>(xspre, ln_w, ln_b, xsh);

  // projections
  {
    dim3 blk(32, 8);
    dim3 gKV((BATCH * NS / 16) / 8, CDIM / (16 * GEMM_NB));
    gemm_f16_wmma<<<gKV, blk, 0, stream>>>(xsh, wkh, nullptr, nullptr, Kh,
                                           BATCH * NS, CDIM, CDIM);
    gemm_f16_wmma<<<gKV, blk, 0, stream>>>(xsh, wvh, nullptr, nullptr, Vh,
                                           BATCH * NS, CDIM, CDIM);
    dim3 gQ((BATCH * NTOK / 16) / 8, CDIM / (16 * GEMM_NB));
    gemm_f16_wmma<<<gQ, blk, 0, stream>>>(xh, wqh, nullptr, nullptr, Qh,
                                          BATCH * NTOK, CDIM, CDIM);
  }

  // V swizzle for attention phase-3 operand layout
  vswz_kernel<<<nblk((size_t)BATCH * NS * CDIM, 256), 256, 0, stream>>>(Vh, Vsw);

  // attention
  {
    dim3 blk(32), grd(NTOK / 16, BATCH * NHEAD);
    attn_kernel<<<grd, blk, 0, stream>>>(Qh, Kh, Vsw, conf, ctxh);
  }

  // output projection (+bias) -> f32 d_out
  {
    dim3 blk(32, 8), grd((BATCH * NTOK / 16) / 8, CDIM / (16 * GEMM_NB));
    gemm_f16_wmma<<<grd, blk, 0, stream>>>(ctxh, wph, bp, (float*)d_out, nullptr,
                                           BATCH * NTOK, CDIM, CDIM);
  }
  (void)in_sizes; (void)n_in; (void)out_size; (void)ws_size;
}
